// SwinBlock2d_9749575762370
// MI455X (gfx1250) — matrix operand order repaired and checked by reading, not compiled, not run on hardware
//
#include <hip/hip_runtime.h>
#include <hip/hip_bf16.h>

// ---------------------------------------------------------------------------
// Swin block for MI455X (gfx1250, wave32, WMMA bf16 path)
// 32x64 register-blocked GEMMs: 8 WMMA per K-step, 12 b128 loads, no spills.
// ---------------------------------------------------------------------------

typedef __attribute__((ext_vector_type(16))) __bf16 v16bf;
typedef __attribute__((ext_vector_type(8)))  __bf16 v8bf;
typedef __attribute__((ext_vector_type(8)))  float  v8f;

union F8 { v8f v; float f[8]; };

__device__ __forceinline__ v8f vzero8() {
  v8f z = {0.f, 0.f, 0.f, 0.f, 0.f, 0.f, 0.f, 0.f};
  return z;
}

__device__ __forceinline__ v8f wmma_bf16(v16bf a, v16bf b, v8f c) {
  // D = A(16x32) * B(32x16) + C, f32 accumulate
  return __builtin_amdgcn_wmma_f32_16x16x32_bf16(false, a, false, b, (short)0, c,
                                                 false, false);
}

// A-operand load (16x32 bf16, row-major src, ld in elements).
// lane L: row = L%16; elements 0..15 = K base..base+15,
// base = (L<16)?0:16  (contiguous order, matches B operands).
__device__ __forceinline__ v16bf load_a16x32(const __bf16* p0, int ld, int lane) {
  const __bf16* p = p0 + (lane & 15) * ld + ((lane & 16) ? 16 : 0);
  v16bf a;
  ((v8bf*)&a)[0] = *(const v8bf*)(p);
  ((v8bf*)&a)[1] = *(const v8bf*)(p + 8);
  return a;
}

// B-operand load from an [N][K] row-major buffer (K contiguous per column-row).
// lane L: column = L%16; elements 0..15 = K kb..kb+15, kb = (L<16)?0:16.
__device__ __forceinline__ v16bf load_b32x16(const __bf16* p0, int ld, int lane) {
  const __bf16* p = p0 + (lane & 15) * ld + ((lane & 16) ? 16 : 0);
  return *(const v16bf*)(p);
}

// 32(M) x 64(N) register-blocked accumulation over K = Kt*32.
// acc[mi*4+j] is the (mi,j) 16x16 f32 tile (mi in 0..1, j in 0..3).
// Per K-step: 4 B-tile loads (pre-packed contiguous), 2 A-tile loads, 8 WMMAs.
__device__ __forceinline__ void gemm_acc_32x64(const __bf16* __restrict__ Arow,
                                               int lda,
                                               const __bf16* __restrict__ Bp,
                                               int Kt, int ntile0, int lane,
                                               v8f* acc) {
  for (int kt = 0; kt < Kt; ++kt) {
    v16bf b[4];
#pragma unroll
    for (int j = 0; j < 4; ++j)
      b[j] = *(const v16bf*)(Bp + ((long)((ntile0 + j) * Kt + kt) * 32 + lane) * 16);
#pragma unroll
    for (int mi = 0; mi < 2; ++mi) {
      v16bf a = load_a16x32(Arow + (long)mi * 16 * lda + kt * 32, lda, lane);
#pragma unroll
      for (int j = 0; j < 4; ++j)
        acc[mi * 4 + j] = wmma_bf16(a, b[j], acc[mi * 4 + j]);
    }
  }
}

// ---------------------------------------------------------------------------
// Weight packing: W[K][N] f32 row-major -> bf16 tiles in B-operand lane order.
// packed[((ntile*Kt + ktile)*32 + lane)*16 + i] = W[ktile*32 + kb + i][ntile*16 + lane%16]
// ---------------------------------------------------------------------------
__global__ void pack_weight(const float* __restrict__ W, __bf16* __restrict__ P,
                            int K, int N) {
  int tid = blockIdx.x * blockDim.x + threadIdx.x;
  int Kt = K >> 5, Nt = N >> 4;
  int total = Nt * Kt * 32;
  if (tid >= total) return;
  int lane = tid & 31, tile = tid >> 5;
  int kt = tile % Kt, nt = tile / Kt;
  int kb = (lane & 16) ? 16 : 0;
  int n = nt * 16 + (lane & 15);
  const float* src = W + (long)(kt * 32 + kb) * N + n;
  __bf16* dst = P + (long)tid * 16;
#pragma unroll
  for (int i = 0; i < 16; ++i) dst[i] = (__bf16)src[(long)i * N];
}

__global__ void fill_zero_u4(uint4* p, long n) {
  long i = (long)blockIdx.x * blockDim.x + threadIdx.x;
  long stride = (long)gridDim.x * blockDim.x;
  uint4 z = {0u, 0u, 0u, 0u};
  for (; i < n; i += stride) p[i] = z;
}

// ---------------------------------------------------------------------------
// LN1 + shift roll(-3,-3) + 7x7 window partition, f32 -> bf16.
// One block (128 thr) per destination pixel; window rows padded 49->64 (pre-zeroed).
// ---------------------------------------------------------------------------
__global__ void ln1_shift_window(const float* __restrict__ x,
                                 const float* __restrict__ gamma,
                                 const float* __restrict__ beta,
                                 __bf16* __restrict__ Awin) {
  int p = blockIdx.x;  // (b, hd, wd) destination pixel
  int bb = p / 3136, r = p % 3136;
  int hd = r / 56, wd = r % 56;
  int hs = (hd + 3) % 56, wsd = (wd + 3) % 56;  // roll(-3): dst[i] = src[(i+3)%56]
  int tid = threadIdx.x;
  __shared__ float red[128];
  float v[3];
#pragma unroll
  for (int i = 0; i < 3; ++i) {
    int c = tid + i * 128;
    v[i] = x[(((long)bb * 384 + c) * 56 + hs) * 56 + wsd];
  }
  red[tid] = v[0] + v[1] + v[2];
  __syncthreads();
  for (int o = 64; o > 0; o >>= 1) {
    if (tid < o) red[tid] += red[tid + o];
    __syncthreads();
  }
  float mean = red[0] * (1.0f / 384.0f);
  __syncthreads();
  float d0 = v[0] - mean, d1 = v[1] - mean, d2 = v[2] - mean;
  red[tid] = d0 * d0 + d1 * d1 + d2 * d2;
  __syncthreads();
  for (int o = 64; o > 0; o >>= 1) {
    if (tid < o) red[tid] += red[tid + o];
    __syncthreads();
  }
  float rstd = rsqrtf(red[0] * (1.0f / 384.0f) + 1e-5f);
  int win = (bb * 8 + hd / 7) * 8 + wd / 7;
  int tok = (hd % 7) * 7 + (wd % 7);
  __bf16* dst = Awin + ((long)win * 64 + tok) * 384;
#pragma unroll
  for (int i = 0; i < 3; ++i) {
    int c = tid + i * 128;
    dst[c] = (__bf16)((v[i] - mean) * rstd * gamma[c] + beta[c]);
  }
}

// ---------------------------------------------------------------------------
// LN2 over contiguous rows of xl [50176][384] f32 -> bf16
// ---------------------------------------------------------------------------
__global__ void ln2_rows(const float* __restrict__ xl,
                         const float* __restrict__ gamma,
                         const float* __restrict__ beta,
                         __bf16* __restrict__ A2) {
  int p = blockIdx.x;
  int tid = threadIdx.x;
  __shared__ float red[128];
  const float* src = xl + (long)p * 384;
  float v[3];
#pragma unroll
  for (int i = 0; i < 3; ++i) v[i] = src[tid + i * 128];
  red[tid] = v[0] + v[1] + v[2];
  __syncthreads();
  for (int o = 64; o > 0; o >>= 1) {
    if (tid < o) red[tid] += red[tid + o];
    __syncthreads();
  }
  float mean = red[0] * (1.0f / 384.0f);
  __syncthreads();
  float d0 = v[0] - mean, d1 = v[1] - mean, d2 = v[2] - mean;
  red[tid] = d0 * d0 + d1 * d1 + d2 * d2;
  __syncthreads();
  for (int o = 64; o > 0; o >>= 1) {
    if (tid < o) red[tid] += red[tid + o];
    __syncthreads();
  }
  float rstd = rsqrtf(red[0] * (1.0f / 384.0f) + 1e-5f);
  __bf16* dst = A2 + (long)p * 384;
#pragma unroll
  for (int i = 0; i < 3; ++i) {
    int c = tid + i * 128;
    dst[c] = (__bf16)((v[i] - mean) * rstd * gamma[c] + beta[c]);
  }
}

// ---------------------------------------------------------------------------
// QKV GEMM: A[65536 x 384] bf16 @ packed(384x1152) + bias, 32x64 wave tiles.
// Outputs: q,k as [win*12+head][64 tok][32 d]; v transposed [..][32 d][64 tok].
// ---------------------------------------------------------------------------
__global__ void __launch_bounds__(256, 1)
qkv_gemm(const __bf16* __restrict__ A, const __bf16* __restrict__ Bp,
         const float* __restrict__ bias, __bf16* __restrict__ q,
         __bf16* __restrict__ k, __bf16* __restrict__ vT) {
  int wid = blockIdx.x * 8 + (threadIdx.x >> 5);
  int lane = threadIdx.x & 31;
  int mblk = wid / 18, ns = wid % 18;
  const __bf16* Arow = A + (long)mblk * 32 * 384;
  v8f acc[8];
#pragma unroll
  for (int i = 0; i < 8; ++i) acc[i] = vzero8();
  gemm_acc_32x64(Arow, 384, Bp, 12, ns * 4, lane, acc);

  int nlo = lane & 15;
  int mofs = (lane & 16) ? 8 : 0;
#pragma unroll
  for (int j = 0; j < 4; ++j) {
    int ng = (ns * 4 + j) * 16 + nlo;
    int which = ng / 384, rem = ng % 384;
    int head = rem >> 5, d = rem & 31;
    float bb = bias[ng];
#pragma unroll
    for (int mi = 0; mi < 2; ++mi) {
      F8 u; u.v = acc[mi * 4 + j];
#pragma unroll
      for (int vv = 0; vv < 8; ++vv) {
        int mg = mblk * 32 + mi * 16 + mofs + vv;
        int win = mg >> 6, tok = mg & 63;
        __bf16 val = (__bf16)(u.f[vv] + bb);
        long base = (long)win * 12 + head;
        if (which == 0)      q[(base * 64 + tok) * 32 + d] = val;
        else if (which == 1) k[(base * 64 + tok) * 32 + d] = val;
        else                 vT[(base * 32 + d) * 64 + tok] = val;
      }
    }
  }
}

// ---------------------------------------------------------------------------
// Attention per (window, head): S = q k^T * scale (16 WMMA), LDS softmax over
// 49 valid tokens, O = P v (16 WMMA, P staged bf16 in LDS).
// ---------------------------------------------------------------------------
__global__ void attn_win(const __bf16* __restrict__ q, const __bf16* __restrict__ k,
                         const __bf16* __restrict__ vT, __bf16* __restrict__ out) {
  int wh = blockIdx.x;
  int win = wh / 12, head = wh % 12;
  int lane = threadIdx.x;
  const __bf16* qp = q + (long)wh * 64 * 32;
  const __bf16* kp = k + (long)wh * 64 * 32;
  const __bf16* vp = vT + (long)wh * 32 * 64;
  __shared__ float S[64 * 64];
  __shared__ __bf16 P[64 * 64];
  const float scale = 0.17677669529663687f;  // 1/sqrt(32)

#pragma unroll
  for (int mi = 0; mi < 4; ++mi) {
    v16bf a = load_a16x32(qp + mi * 16 * 32, 32, lane);
#pragma unroll
    for (int ni = 0; ni < 4; ++ni) {
      // B = K^T: B[k=d][n=tok] = kmem[tok][d] -> contiguous per lane
      v16bf bv = load_b32x16(kp + ni * 16 * 32, 32, lane);
      v8f c = vzero8();
      c = wmma_bf16(a, bv, c);
      F8 u; u.v = c;
      int n = ni * 16 + (lane & 15);
      int mo = (lane & 16) ? 8 : 0;
#pragma unroll
      for (int vv = 0; vv < 8; ++vv)
        S[(mi * 16 + mo + vv) * 64 + n] = u.f[vv] * scale;
    }
  }
  __syncthreads();

  // softmax: each lane owns rows (lane) and (lane+32); mask tokens >= 49
#pragma unroll
  for (int rr = 0; rr < 2; ++rr) {
    int row = lane + rr * 32;
    float mx = -1e30f;
    for (int j = 0; j < 49; ++j) mx = fmaxf(mx, S[row * 64 + j]);
    float ssum = 0.0f;
    for (int j = 0; j < 49; ++j) {
      float e = __expf(S[row * 64 + j] - mx);
      S[row * 64 + j] = e;
      ssum += e;
    }
    float inv = 1.0f / ssum;
    for (int j = 0; j < 49; ++j) P[row * 64 + j] = (__bf16)(S[row * 64 + j] * inv);
    for (int j = 49; j < 64; ++j) P[row * 64 + j] = (__bf16)0.0f;
  }
  __syncthreads();

#pragma unroll
  for (int mi = 0; mi < 4; ++mi) {
    v8f o0 = vzero8(), o1 = vzero8();
#pragma unroll
    for (int kt = 0; kt < 2; ++kt) {
      v16bf a = load_a16x32((const __bf16*)P + mi * 16 * 64 + kt * 32, 64, lane);
      // B = V: B[k=tok][n=d] = vT[d][tok] -> contiguous per lane
      v16bf b0 = load_b32x16(vp + 0 * 64 + kt * 32, 64, lane);
      v16bf b1 = load_b32x16(vp + 16 * 64 + kt * 32, 64, lane);
      o0 = wmma_bf16(a, b0, o0);
      o1 = wmma_bf16(a, b1, o1);
    }
    int mo = (lane & 16) ? 8 : 0;
    int nlo = lane & 15;
    F8 u0, u1; u0.v = o0; u1.v = o1;
#pragma unroll
    for (int vv = 0; vv < 8; ++vv) {
      int tok = mi * 16 + mo + vv;
      long rowbase = ((long)win * 64 + tok) * 384 + head * 32;
      out[rowbase + nlo] = (__bf16)u0.f[vv];
      out[rowbase + 16 + nlo] = (__bf16)u1.f[vv];
    }
  }
}

// ---------------------------------------------------------------------------
// Proj GEMM + window-merge + reverse shift + residual -> xl f32 [B*H*W][384]
// ---------------------------------------------------------------------------
__global__ void __launch_bounds__(256, 1)
proj_gemm(const __bf16* __restrict__ A, const __bf16* __restrict__ Bp,
          const float* __restrict__ bias, const float* __restrict__ x,
          float* __restrict__ xl) {
  int wid = blockIdx.x * 8 + (threadIdx.x >> 5);
  int lane = threadIdx.x & 31;
  int mblk = wid / 6, ns = wid % 6;
  const __bf16* Arow = A + (long)mblk * 32 * 384;
  v8f acc[8];
#pragma unroll
  for (int i = 0; i < 8; ++i) acc[i] = vzero8();
  gemm_acc_32x64(Arow, 384, Bp, 12, ns * 4, lane, acc);

  int nlo = lane & 15, mofs = (lane & 16) ? 8 : 0;
#pragma unroll
  for (int j = 0; j < 4; ++j) {
    int c = (ns * 4 + j) * 16 + nlo;
    float bb = bias[c];
#pragma unroll
    for (int mi = 0; mi < 2; ++mi) {
      F8 u; u.v = acc[mi * 4 + j];
#pragma unroll
      for (int vv = 0; vv < 8; ++vv) {
        int mg = mblk * 32 + mi * 16 + mofs + vv;
        int win = mg >> 6, tok = mg & 63;
        if (tok >= 49) continue;  // padded window rows
        int bbi = win >> 6;
        int hb = (win & 63) >> 3, wb = win & 7;
        int hd = hb * 7 + tok / 7, wd = wb * 7 + tok % 7;
        int hf = (hd + 3) % 56, wf = (wd + 3) % 56;  // roll(+3)
        float res = x[(((long)bbi * 384 + c) * 56 + hf) * 56 + wf];
        xl[(((long)bbi * 56 + hf) * 56 + wf) * 384 + c] = res + u.f[vv] + bb;
      }
    }
  }
}

// ---------------------------------------------------------------------------
// FC1 GEMM + exact GELU -> H bf16 [50176][1536]
// ---------------------------------------------------------------------------
__global__ void __launch_bounds__(256, 1)
fc1_gemm(const __bf16* __restrict__ A, const __bf16* __restrict__ Bp,
         const float* __restrict__ bias, __bf16* __restrict__ Hb) {
  int wid = blockIdx.x * 8 + (threadIdx.x >> 5);
  int lane = threadIdx.x & 31;
  int mblk = wid / 24, ns = wid % 24;
  const __bf16* Arow = A + (long)mblk * 32 * 384;
  v8f acc[8];
#pragma unroll
  for (int i = 0; i < 8; ++i) acc[i] = vzero8();
  gemm_acc_32x64(Arow, 384, Bp, 12, ns * 4, lane, acc);

  int nlo = lane & 15, mofs = (lane & 16) ? 8 : 0;
#pragma unroll
  for (int j = 0; j < 4; ++j) {
    int c = (ns * 4 + j) * 16 + nlo;
    float bb = bias[c];
#pragma unroll
    for (int mi = 0; mi < 2; ++mi) {
      F8 u; u.v = acc[mi * 4 + j];
#pragma unroll
      for (int vv = 0; vv < 8; ++vv) {
        int mg = mblk * 32 + mi * 16 + mofs + vv;
        float val = u.f[vv] + bb;
        float g = 0.5f * val * (1.0f + erff(val * 0.70710678118654752f));
        Hb[(long)mg * 1536 + c] = (__bf16)g;
      }
    }
  }
}

// ---------------------------------------------------------------------------
// FC2 GEMM + residual + NHWC->NCHW transpose store -> d_out f32
// ---------------------------------------------------------------------------
__global__ void __launch_bounds__(256, 1)
fc2_gemm(const __bf16* __restrict__ A, const __bf16* __restrict__ Bp,
         const float* __restrict__ bias, const float* __restrict__ xl,
         float* __restrict__ out) {
  int wid = blockIdx.x * 8 + (threadIdx.x >> 5);
  int lane = threadIdx.x & 31;
  int mblk = wid / 6, ns = wid % 6;
  const __bf16* Arow = A + (long)mblk * 32 * 1536;
  v8f acc[8];
#pragma unroll
  for (int i = 0; i < 8; ++i) acc[i] = vzero8();
  gemm_acc_32x64(Arow, 1536, Bp, 48, ns * 4, lane, acc);

  int nlo = lane & 15, mofs = (lane & 16) ? 8 : 0;
#pragma unroll
  for (int j = 0; j < 4; ++j) {
    int c = (ns * 4 + j) * 16 + nlo;
    float bb = bias[c];
#pragma unroll
    for (int mi = 0; mi < 2; ++mi) {
      F8 u; u.v = acc[mi * 4 + j];
#pragma unroll
      for (int vv = 0; vv < 8; ++vv) {
        int mg = mblk * 32 + mi * 16 + mofs + vv;
        float val = u.f[vv] + bb + xl[(long)mg * 384 + c];
        int bbi = mg / 3136, rem = mg % 3136;
        int h = rem / 56, w = rem % 56;
        out[(((long)bbi * 384 + c) * 56 + h) * 56 + w] = val;
      }
    }
  }
}

// ---------------------------------------------------------------------------
// Host launcher. Workspace layout (bytes), ~324 MB with aliasing:
//   [0)            packed qkv_w   (884736)
//   [884736)       packed proj_w  (294912)
//   [1179648)      packed fc1_w   (1179648)
//   [2359296)      packed fc2_w   (1179648)
//   [3538944)      Awin  bf16 [1024][64][384]  (50331648)  -> reused as attn_out
//   [53870592)     q / k / vT  (3 x 50331648)              -> reused as H bf16
//   [208011264)    xl f32 [50176][384] (77070336)
//   [285081600)    A2 bf16 [50176][384] (38535168)         end = 323616768
// ---------------------------------------------------------------------------
extern "C" void kernel_launch(void* const* d_in, const int* in_sizes, int n_in,
                              void* d_out, int out_size, void* d_ws, size_t ws_size,
                              hipStream_t stream) {
  (void)in_sizes; (void)n_in; (void)out_size; (void)ws_size;
  const float* x      = (const float*)d_in[0];
  const float* n1w    = (const float*)d_in[1];
  const float* n1b    = (const float*)d_in[2];
  const float* qkv_w  = (const float*)d_in[3];
  const float* qkv_b  = (const float*)d_in[4];
  const float* proj_w = (const float*)d_in[5];
  const float* proj_b = (const float*)d_in[6];
  const float* n2w    = (const float*)d_in[7];
  const float* n2b    = (const float*)d_in[8];
  const float* fc1_w  = (const float*)d_in[9];
  const float* fc1_b  = (const float*)d_in[10];
  const float* fc2_w  = (const float*)d_in[11];
  const float* fc2_b  = (const float*)d_in[12];

  char* ws = (char*)d_ws;
  __bf16* qkvp = (__bf16*)(ws + 0);
  __bf16* projp = (__bf16*)(ws + 884736);
  __bf16* fc1p = (__bf16*)(ws + 1179648);
  __bf16* fc2p = (__bf16*)(ws + 2359296);
  __bf16* Awin = (__bf16*)(ws + 3538944);        // also attention output
  __bf16* q    = (__bf16*)(ws + 53870592);
  __bf16* kk   = (__bf16*)(ws + 104202240);
  __bf16* vT   = (__bf16*)(ws + 154533888);
  __bf16* Hb   = (__bf16*)(ws + 53870592);       // aliases q/k/vT (after attn)
  float*  xl   = (float*)(ws + 208011264);
  __bf16* A2   = (__bf16*)(ws + 285081600);
  float*  out  = (float*)d_out;

  pack_weight<<<108, 256, 0, stream>>>(qkv_w, qkvp, 384, 1152);
  pack_weight<<<36, 256, 0, stream>>>(proj_w, projp, 384, 384);
  pack_weight<<<144, 256, 0, stream>>>(fc1_w, fc1p, 384, 1536);
  pack_weight<<<144, 256, 0, stream>>>(fc2_w, fc2p, 1536, 384);

  fill_zero_u4<<<4096, 256, 0, stream>>>((uint4*)Awin, 50331648L / 16);
  ln1_shift_window<<<50176, 128, 0, stream>>>(x, n1w, n1b, Awin);
  // 65536 rows / 32 = 2048 mblks; 18 strips -> 36864 waves / 8
  qkv_gemm<<<4608, 256, 0, stream>>>(Awin, qkvp, qkv_b, q, kk, vT);
  attn_win<<<12288, 32, 0, stream>>>(q, kk, vT, Awin);
  // 2048 mblks * 6 strips -> 12288 waves / 8
  proj_gemm<<<1536, 256, 0, stream>>>(Awin, projp, proj_b, x, xl);
  ln2_rows<<<50176, 128, 0, stream>>>(xl, n2w, n2b, A2);
  // 1568 mblks * 24 strips -> 37632 waves / 8
  fc1_gemm<<<4704, 256, 0, stream>>>(A2, fc1p, fc1_b, Hb);
  // 1568 mblks * 6 strips -> 9408 waves / 8
  fc2_gemm<<<1176, 256, 0, stream>>>(Hb, fc2p, fc2_b, xl, out);
}
